// RaftPlusDiclModule_23897198035029
// MI455X (gfx1250) — matrix operand order repaired and checked by reading, not compiled, not run on hardware
//
#include <hip/hip_runtime.h>
#include <hip/hip_bf16.h>

typedef _Float16 half_t;
typedef __attribute__((ext_vector_type(16))) _Float16 v16h;
typedef __attribute__((ext_vector_type(8)))  _Float16 v8h;
typedef __attribute__((ext_vector_type(8)))  float    v8f;

#define NIMG 162              // B * D * D = 2 * 81
#define HF   48
#define WF   64
#define PQF  (HF * WF)        // 3072

// ---------------------------------------------------------------------------
// Weight repack: [Kout][Cin][3][3] f32 -> [Kout][9*Cin] f16, k = (r*3+s)*Cin + c
// ---------------------------------------------------------------------------
__global__ void repack_w3x3(const float* __restrict__ src, half_t* __restrict__ dst,
                            int Kout, int Cin) {
  int idx = blockIdx.x * 256 + threadIdx.x;
  int total = Kout * Cin * 9;
  if (idx >= total) return;
  int o   = idx / (Cin * 9);
  int rem = idx % (Cin * 9);
  int c   = rem / 9;
  int rs  = rem % 9;
  dst[(size_t)o * (9 * Cin) + rs * Cin + c] = (half_t)src[idx];
}

// w5 [32][64][4][4] f32 -> w5r[cls=4][32][256] f16, k = (a*2+b)*64 + c,
// ky = (cls>>1) + 2a, kx = (cls&1) + 2b   (parity-class GEMM A operand)
__global__ void repack_w5(const float* __restrict__ src, half_t* __restrict__ dst) {
  int idx = blockIdx.x * 256 + threadIdx.x;
  if (idx >= 4 * 32 * 256) return;
  int cls = idx / (32 * 256);
  int rem = idx % (32 * 256);
  int o = rem / 256, k = rem % 256;
  int tap = k >> 6, c = k & 63;
  int a = tap >> 1, b = tap & 1;
  int ky = (cls >> 1) + 2 * a, kx = (cls & 1) + 2 * b;
  dst[idx] = (half_t)src[((size_t)(o * 64 + c)) * 16 + ky * 4 + kx];
}

// ---------------------------------------------------------------------------
// Fused displacement gather (bilinear, align_corners, zero pad) + concat.
// Output corr in NHWC f16: [NIMG][48*64][64]  (contiguous 128B per pixel)
// ---------------------------------------------------------------------------
__global__ void build_corr(const float* __restrict__ f1,      // [2][32][48][64]
                           const float* __restrict__ f2,      // [2][32][h2][w2]
                           const float* __restrict__ coords,  // [2][2][48][64]
                           half_t* __restrict__ corr,
                           int level, int h2, int w2) {
  int idx = blockIdx.x * 256 + threadIdx.x;
  if (idx >= NIMG * PQF) return;
  int n  = idx / PQF;
  int gp = idx % PQF;
  int b  = n / 81;
  int dd = n % 81;
  float dx = (float)(dd / 9 - 4);
  float dy = (float)(dd % 9 - 4);
  float scale = 1.0f / (float)(1 << level);
  float cx = coords[(size_t)(b * 2 + 0) * PQF + gp];
  float cy = coords[(size_t)(b * 2 + 1) * PQF + gp];
  float gx = cx * scale + dx;
  float gy = cy * scale + dy;
  float x0f = floorf(gx), y0f = floorf(gy);
  int   x0 = (int)x0f,    y0 = (int)y0f;
  float wx = gx - x0f,    wy = gy - y0f;
  float w00 = (1.f - wx) * (1.f - wy), w10 = wx * (1.f - wy);
  float w01 = (1.f - wx) * wy,         w11 = wx * wy;
  bool vx0 = (x0 >= 0) && (x0 <= w2 - 1);
  bool vx1 = (x0 + 1 >= 0) && (x0 + 1 <= w2 - 1);
  bool vy0 = (y0 >= 0) && (y0 <= h2 - 1);
  bool vy1 = (y0 + 1 >= 0) && (y0 + 1 <= h2 - 1);
  w00 *= (float)(vx0 && vy0); w10 *= (float)(vx1 && vy0);
  w01 *= (float)(vx0 && vy1); w11 *= (float)(vx1 && vy1);
  int xc0 = min(max(x0, 0), w2 - 1),  xc1 = min(max(x0 + 1, 0), w2 - 1);
  int yc0 = min(max(y0, 0), h2 - 1),  yc1 = min(max(y0 + 1, 0), h2 - 1);
  const float* f2b = f2 + (size_t)b * 32 * h2 * w2;

  half_t tmp[64];
#pragma unroll
  for (int c = 0; c < 32; ++c) {
    tmp[c] = (half_t)f1[(size_t)(b * 32 + c) * PQF + gp];
    const float* fc = f2b + (size_t)c * h2 * w2;
    float v = w00 * fc[yc0 * w2 + xc0] + w10 * fc[yc0 * w2 + xc1]
            + w01 * fc[yc1 * w2 + xc0] + w11 * fc[yc1 * w2 + xc1];
    tmp[32 + c] = (half_t)v;
  }
  half_t* row = corr + ((size_t)n * PQF + gp) * 64;
#pragma unroll
  for (int v8 = 0; v8 < 8; ++v8)
    *(v8h*)&row[v8 * 8] = *(const v8h*)&tmp[v8 * 8];
}

// ---------------------------------------------------------------------------
// WMMA fragment helper (ISA 7.12.2 layouts, wave32)
// ---------------------------------------------------------------------------
__device__ __forceinline__ v16h cat8(v8h lo, v8h hi) {
  union { v16h v; v8h h[2]; } u;
  u.h[0] = lo; u.h[1] = hi;
  return u.v;
}

// ---------------------------------------------------------------------------
// Implicit-GEMM 3x3 conv (+bias +ReLU), NHWC f16 in/out, f32 WMMA accumulate.
// act [N][Hin*Win][Cin], wgt [Kout][9*Cin], out [N][P*Q][Kout]
// Block: 128 thr (4 waves). Tile 32 outch x 128 pixels; per-wave 32x32 quadrant
// (2 A frags x 2 B frags -> 4 wmma / k-step). Double-buffered LDS staging,
// one barrier per k-step. Requires Cin%32==0, P*Q%128==0, Kout%32==0.
// ---------------------------------------------------------------------------
template<int STRIDE>
__global__ __launch_bounds__(128)
void conv3x3_wmma(const half_t* __restrict__ act, const half_t* __restrict__ wgt,
                  const float* __restrict__ bias, half_t* __restrict__ out,
                  int Cin, int Hin, int Win, int Kout, int P, int Q) {
  const int K9    = 9 * Cin;
  const int PQ    = P * Q;
  const int n     = blockIdx.z;
  const int kout0 = blockIdx.y * 32;
  const int p0    = blockIdx.x * 128;
  const int t     = threadIdx.x;
  const int lane  = t & 31;
  const int ncol0 = (t >> 5) * 32;        // wave's N offset (4 waves x 32)

  __shared__ __align__(16) half_t As[2][32][40];
  __shared__ __align__(16) half_t Bs[2][128][40];

  v8f acc00 = {0.f,0.f,0.f,0.f,0.f,0.f,0.f,0.f};
  v8f acc01 = acc00, acc10 = acc00, acc11 = acc00;

  // A staging: thread -> 8 contiguous k of row mA
  const int mA  = t >> 2;
  const int kA8 = (t & 3) * 8;
  const half_t* wrow = wgt + (size_t)(kout0 + mA) * K9 + kA8;

  // B staging: one thread per pixel, 32 contiguous halfs (NHWC!)
  const int pB  = t;
  const int gpB = p0 + pB;
  const int oyB = gpB / Q;
  const int oxB = gpB % Q;
  const half_t* actn = act + (size_t)n * Hin * Win * Cin;

  const v8h vzero = {(_Float16)0,(_Float16)0,(_Float16)0,(_Float16)0,
                     (_Float16)0,(_Float16)0,(_Float16)0,(_Float16)0};
  v8h aReg, bReg0, bReg1, bReg2, bReg3;

  auto loadAB = [&](int kk) {
    aReg = *(const v8h*)(wrow + kk);
    int rs = kk / Cin;                 // constant per 32-chunk (Cin%32==0)
    int c0 = kk - rs * Cin;
    int r = rs / 3, s = rs % 3;
    int iy = oyB * STRIDE + r - 1;
    int ix = oxB * STRIDE + s - 1;
    if ((iy >= 0) && (iy < Hin) && (ix >= 0) && (ix < Win)) {
      const half_t* ap = actn + ((size_t)iy * Win + ix) * Cin + c0;
      bReg0 = *(const v8h*)(ap);
      bReg1 = *(const v8h*)(ap + 8);
      bReg2 = *(const v8h*)(ap + 16);
      bReg3 = *(const v8h*)(ap + 24);
    } else {
      bReg0 = vzero; bReg1 = vzero; bReg2 = vzero; bReg3 = vzero;
    }
  };
  auto storeAB = [&](int buf) {
    *(v8h*)&As[buf][mA][kA8] = aReg;
    *(v8h*)&Bs[buf][pB][0]  = bReg0;
    *(v8h*)&Bs[buf][pB][8]  = bReg1;
    *(v8h*)&Bs[buf][pB][16] = bReg2;
    *(v8h*)&Bs[buf][pB][24] = bReg3;
  };

  loadAB(0);
  storeAB(0);
  __syncthreads();

  int cur = 0;
  for (int kk = 0; kk < K9; kk += 32) {
    int nxt = kk + 32;
    if (nxt < K9) loadAB(nxt);         // global loads in flight during wmma
    // fragments (ds_load_b128 x8, conflict-free via 8-half row pad)
    const int Ml = lane & 15;
    const int kA = (lane >> 4) * 8;
    v16h a0 = cat8(*(const v8h*)&As[cur][Ml][kA],      *(const v8h*)&As[cur][Ml][kA + 16]);
    v16h a1 = cat8(*(const v8h*)&As[cur][Ml + 16][kA], *(const v8h*)&As[cur][Ml + 16][kA + 16]);
    const int kB = (lane >> 4) * 8;
    const int N0 = ncol0 + Ml;
    v16h b0 = cat8(*(const v8h*)&Bs[cur][N0][kB],      *(const v8h*)&Bs[cur][N0][kB + 16]);
    v16h b1 = cat8(*(const v8h*)&Bs[cur][N0 + 16][kB], *(const v8h*)&Bs[cur][N0 + 16][kB + 16]);
    acc00 = __builtin_amdgcn_wmma_f32_16x16x32_f16(false, a0, false, b0, (short)0, acc00, false, false);
    acc01 = __builtin_amdgcn_wmma_f32_16x16x32_f16(false, a0, false, b1, (short)0, acc01, false, false);
    acc10 = __builtin_amdgcn_wmma_f32_16x16x32_f16(false, a1, false, b0, (short)0, acc10, false, false);
    acc11 = __builtin_amdgcn_wmma_f32_16x16x32_f16(false, a1, false, b1, (short)0, acc11, false, false);
    if (nxt < K9) storeAB(cur ^ 1);    // opposite buffer: no read/write overlap
    __syncthreads();                   // single barrier per k-step
    cur ^= 1;
  }

  // epilogue: lane holds 8 consecutive kout for one pixel -> b128 NHWC store
  const int rbase = 8 * (lane >> 4);
  const int ncol  = lane & 15;
  auto emit = [&](v8f acc, int mbase, int noff) {
    int koutb = kout0 + mbase + rbase;
    v8h o;
#pragma unroll
    for (int r = 0; r < 8; ++r)
      o[r] = (half_t)fmaxf(acc[r] + bias[koutb + r], 0.f);
    int gp = p0 + ncol0 + noff + ncol;
    *(v8h*)&out[((size_t)n * PQ + gp) * Kout + koutb] = o;
  };
  emit(acc00, 0, 0); emit(acc01, 0, 16); emit(acc10, 16, 0); emit(acc11, 16, 16);
}

// ---------------------------------------------------------------------------
// ConvTranspose2d(64->32, k4, s2, p1) + bias + ReLU as parity-class GEMM.
// Kdim = 4 taps * 64 ch = 256. act NHWC [N][24*32][64], w5r [4][32][256],
// out NHWC [N][48*64][32]. Block per (128-pixel class tile, n*4+cls).
// ---------------------------------------------------------------------------
__global__ __launch_bounds__(128)
void deconv4x4_wmma(const half_t* __restrict__ act, const half_t* __restrict__ w5r,
                    const float* __restrict__ bias, half_t* __restrict__ out) {
  const int nz  = blockIdx.z;
  const int n   = nz >> 2;
  const int cls = nz & 3;
  const int py  = cls >> 1, px = cls & 1;
  const int p0  = blockIdx.x * 128;      // class pixels (768 per class)
  const int t    = threadIdx.x;
  const int lane = t & 31;
  const int ncol0 = (t >> 5) * 32;

  __shared__ __align__(16) half_t As[2][32][40];
  __shared__ __align__(16) half_t Bs[2][128][40];

  v8f acc00 = {0.f,0.f,0.f,0.f,0.f,0.f,0.f,0.f};
  v8f acc01 = acc00, acc10 = acc00, acc11 = acc00;

  const int mA  = t >> 2;
  const int kA8 = (t & 3) * 8;
  const half_t* wrow = w5r + ((size_t)cls * 32 + mA) * 256 + kA8;

  const int pB  = t;
  const int pcB = p0 + pB;
  const int tyB = pcB >> 5, txB = pcB & 31;
  const half_t* actn = act + (size_t)n * 24 * 32 * 64;

  const v8h vzero = {(_Float16)0,(_Float16)0,(_Float16)0,(_Float16)0,
                     (_Float16)0,(_Float16)0,(_Float16)0,(_Float16)0};
  v8h aReg, bReg0, bReg1, bReg2, bReg3;

  auto loadAB = [&](int kk) {
    aReg = *(const v8h*)(wrow + kk);
    int tap = kk >> 6;                  // constant per 32-chunk
    int a_ = tap >> 1, b_ = tap & 1;
    int iy = tyB + py + a_ - 1;
    int ix = txB + px + b_ - 1;
    int c0 = kk & 63;
    if ((iy >= 0) && (iy < 24) && (ix >= 0) && (ix < 32)) {
      const half_t* ap = actn + ((size_t)iy * 32 + ix) * 64 + c0;
      bReg0 = *(const v8h*)(ap);
      bReg1 = *(const v8h*)(ap + 8);
      bReg2 = *(const v8h*)(ap + 16);
      bReg3 = *(const v8h*)(ap + 24);
    } else {
      bReg0 = vzero; bReg1 = vzero; bReg2 = vzero; bReg3 = vzero;
    }
  };
  auto storeAB = [&](int buf) {
    *(v8h*)&As[buf][mA][kA8] = aReg;
    *(v8h*)&Bs[buf][pB][0]  = bReg0;
    *(v8h*)&Bs[buf][pB][8]  = bReg1;
    *(v8h*)&Bs[buf][pB][16] = bReg2;
    *(v8h*)&Bs[buf][pB][24] = bReg3;
  };

  loadAB(0);
  storeAB(0);
  __syncthreads();

  int cur = 0;
  for (int kk = 0; kk < 256; kk += 32) {
    int nxt = kk + 32;
    if (nxt < 256) loadAB(nxt);
    const int Ml = lane & 15;
    const int kA = (lane >> 4) * 8;
    v16h a0 = cat8(*(const v8h*)&As[cur][Ml][kA],      *(const v8h*)&As[cur][Ml][kA + 16]);
    v16h a1 = cat8(*(const v8h*)&As[cur][Ml + 16][kA], *(const v8h*)&As[cur][Ml + 16][kA + 16]);
    const int kB = (lane >> 4) * 8;
    const int N0 = ncol0 + Ml;
    v16h b0 = cat8(*(const v8h*)&Bs[cur][N0][kB],      *(const v8h*)&Bs[cur][N0][kB + 16]);
    v16h b1 = cat8(*(const v8h*)&Bs[cur][N0 + 16][kB], *(const v8h*)&Bs[cur][N0 + 16][kB + 16]);
    acc00 = __builtin_amdgcn_wmma_f32_16x16x32_f16(false, a0, false, b0, (short)0, acc00, false, false);
    acc01 = __builtin_amdgcn_wmma_f32_16x16x32_f16(false, a0, false, b1, (short)0, acc01, false, false);
    acc10 = __builtin_amdgcn_wmma_f32_16x16x32_f16(false, a1, false, b0, (short)0, acc10, false, false);
    acc11 = __builtin_amdgcn_wmma_f32_16x16x32_f16(false, a1, false, b1, (short)0, acc11, false, false);
    if (nxt < 256) storeAB(cur ^ 1);
    __syncthreads();
    cur ^= 1;
  }

  const int rbase = 8 * (lane >> 4);
  const int ncol  = lane & 15;
  auto emit = [&](v8f acc, int mbase, int noff) {
    int koutb = mbase + rbase;
    v8h o;
#pragma unroll
    for (int r = 0; r < 8; ++r)
      o[r] = (half_t)fmaxf(acc[r] + bias[koutb + r], 0.f);
    int pc = p0 + ncol0 + noff + ncol;
    int y = 2 * (pc >> 5) + py, x = 2 * (pc & 31) + px;
    *(v8h*)&out[((size_t)n * PQF + (size_t)y * WF + x) * 32 + koutb] = o;
  };
  emit(acc00, 0, 0); emit(acc01, 0, 16); emit(acc10, 16, 0); emit(acc11, 16, 16);
}

// ---------------------------------------------------------------------------
// conv6: 32 -> 1 channel, 3x3 + bias (no relu). act5 NHWC. FLOP-trivial.
// ---------------------------------------------------------------------------
__global__ void conv6_k(const half_t* __restrict__ act5, const float* __restrict__ w6,
                        const float* __restrict__ b6, float* __restrict__ cost) {
  int idx = blockIdx.x * 256 + threadIdx.x;
  if (idx >= NIMG * PQF) return;
  int n = idx / PQF, gp = idx % PQF;
  int y = gp / WF, x = gp % WF;
  const half_t* base = act5 + (size_t)n * PQF * 32;
  float acc = b6[0];
#pragma unroll
  for (int r = 0; r < 3; ++r) {
    int iy = y + r - 1;
    if (iy < 0 || iy >= HF) continue;
#pragma unroll
    for (int s = 0; s < 3; ++s) {
      int ix = x + s - 1;
      if (ix < 0 || ix >= WF) continue;
      const half_t* ap = base + ((size_t)iy * WF + ix) * 32;
      const float*  wp = w6 + (r * 3 + s);
      float sum = 0.f;
#pragma unroll
      for (int c = 0; c < 32; ++c) sum += (float)ap[c] * wp[c * 9];
      acc += sum;
    }
  }
  cost[idx] = acc;
}

// ---------------------------------------------------------------------------
// DAP: out[b][lvl*81+o][gp] = sum_d dapw[o][d] * cost[(b*81+d)][gp]
// ---------------------------------------------------------------------------
__global__ void dap_k(const float* __restrict__ cost, const float* __restrict__ dw,
                      float* __restrict__ out, int level) {
  int idx = blockIdx.x * 256 + threadIdx.x;
  if (idx >= 2 * 81 * PQF) return;
  int gp = idx % PQF;
  int bo = idx / PQF;
  int b = bo / 81, o = bo % 81;
  float acc = 0.f;
  const float* dwo = dw + o * 81;
  const float* cb  = cost + (size_t)b * 81 * PQF + gp;
  for (int d = 0; d < 81; ++d) acc += dwo[d] * cb[(size_t)d * PQF];
  out[((size_t)(b * 324 + level * 81 + o)) * PQF + gp] = acc;
}

// ---------------------------------------------------------------------------
extern "C" void kernel_launch(void* const* d_in, const int* in_sizes, int n_in,
                              void* d_out, int out_size, void* d_ws, size_t ws_size,
                              hipStream_t stream) {
  (void)in_sizes; (void)n_in; (void)out_size; (void)ws_size;
  const float* fmap1[4] = {(const float*)d_in[0], (const float*)d_in[1],
                           (const float*)d_in[2], (const float*)d_in[3]};
  const float* fmap2[4] = {(const float*)d_in[4], (const float*)d_in[5],
                           (const float*)d_in[6], (const float*)d_in[7]};
  const float* coords = (const float*)d_in[8];
  const float* mw1 = (const float*)d_in[9];   const float* mb1 = (const float*)d_in[10];
  const float* mw2 = (const float*)d_in[11];  const float* mb2 = (const float*)d_in[12];
  const float* mw3 = (const float*)d_in[13];  const float* mb3 = (const float*)d_in[14];
  const float* mw4 = (const float*)d_in[15];  const float* mb4 = (const float*)d_in[16];
  const float* mw5 = (const float*)d_in[17];  const float* mb5 = (const float*)d_in[18];
  const float* mw6 = (const float*)d_in[19];  const float* mb6 = (const float*)d_in[20];
  const float* dapw = (const float*)d_in[21];

  char* ws = (char*)d_ws;
  size_t off = 0;
  auto alloc = [&](size_t bytes) {
    void* p = ws + off;
    off = (off + bytes + 255) & ~(size_t)255;
    return p;
  };
  half_t* bufA = (half_t*)alloc((size_t)NIMG * 64 * PQF * 2);   // corr / act2 / act4
  half_t* bufB = (half_t*)alloc((size_t)NIMG * 96 * PQF * 2);   // act1 / act3 / act5
  float*  cost = (float*) alloc((size_t)NIMG * PQF * 4);
  half_t* w1h  = (half_t*)alloc((size_t)96 * 64 * 9 * 2);
  half_t* w2h  = (half_t*)alloc((size_t)128 * 96 * 9 * 2);
  half_t* w3h  = (half_t*)alloc((size_t)128 * 128 * 9 * 2);
  half_t* w4h  = (half_t*)alloc((size_t)64 * 128 * 9 * 2);
  half_t* w5h  = (half_t*)alloc((size_t)4 * 32 * 256 * 2);

  const int NPQ = NIMG * PQF;           // 497664
  for (int lvl = 0; lvl < 4; ++lvl) {
    int h2 = HF >> lvl, w2d = WF >> lvl;
    repack_w3x3<<<(96 * 64 * 9 + 255) / 256, 256, 0, stream>>>(mw1 + (size_t)lvl * 96 * 64 * 9, w1h, 96, 64);
    repack_w3x3<<<(128 * 96 * 9 + 255) / 256, 256, 0, stream>>>(mw2 + (size_t)lvl * 128 * 96 * 9, w2h, 128, 96);
    repack_w3x3<<<(128 * 128 * 9 + 255) / 256, 256, 0, stream>>>(mw3 + (size_t)lvl * 128 * 128 * 9, w3h, 128, 128);
    repack_w3x3<<<(64 * 128 * 9 + 255) / 256, 256, 0, stream>>>(mw4 + (size_t)lvl * 64 * 128 * 9, w4h, 64, 128);
    repack_w5<<<(4 * 32 * 256 + 255) / 256, 256, 0, stream>>>(mw5 + (size_t)lvl * 32 * 64 * 16, w5h);

    build_corr<<<(NPQ + 255) / 256, 256, 0, stream>>>(fmap1[lvl], fmap2[lvl], coords,
                                                      bufA, lvl, h2, w2d);
    conv3x3_wmma<1><<<dim3(PQF / 128, 96 / 32, NIMG), 128, 0, stream>>>(
        bufA, w1h, mb1 + lvl * 96, bufB, 64, 48, 64, 96, 48, 64);
    conv3x3_wmma<2><<<dim3(768 / 128, 128 / 32, NIMG), 128, 0, stream>>>(
        bufB, w2h, mb2 + lvl * 128, bufA, 96, 48, 64, 128, 24, 32);
    conv3x3_wmma<1><<<dim3(768 / 128, 128 / 32, NIMG), 128, 0, stream>>>(
        bufA, w3h, mb3 + lvl * 128, bufB, 128, 24, 32, 128, 24, 32);
    conv3x3_wmma<1><<<dim3(768 / 128, 64 / 32, NIMG), 128, 0, stream>>>(
        bufB, w4h, mb4 + lvl * 64, bufA, 128, 24, 32, 64, 24, 32);
    deconv4x4_wmma<<<dim3(768 / 128, 1, NIMG * 4), 128, 0, stream>>>(
        bufA, w5h, mb5 + lvl * 32, bufB);
    conv6_k<<<(NPQ + 255) / 256, 256, 0, stream>>>(bufB, mw6 + lvl * 288, mb6 + lvl, cost);
    dap_k<<<(2 * 81 * PQF + 255) / 256, 256, 0, stream>>>(cost, dapw + lvl * 81 * 81,
                                                          (float*)d_out, lvl);
  }
}
